// PKM_54752243089430
// MI455X (gfx1250) — matrix operand order repaired and checked by reading, not compiled, not run on hardware
//
#include <hip/hip_runtime.h>

// ---------------- problem constants ----------------
#define NTOK  4096      // B*T
#define DIMX  512       // DIM
#define NQ    2048      // 2*H*DH
#define HH    8
#define DH    128
#define NKEY  256       // NK
#define TOPK  16
#define LN_EPS 1e-5f

typedef __bf16 bf16_t;
typedef __attribute__((ext_vector_type(16))) __bf16 v16bf;
typedef __attribute__((ext_vector_type(8)))  __bf16 v8bf;
typedef __attribute__((ext_vector_type(8)))  float  v8f;

union FragBF { v16bf v; v8bf h[2]; };

// =====================================================================
// Kernel 1: q = x @ Wq^T  (4096x512 @ 512x2048), fused LayerNorm over
// each 128-wide segment, output bf16. 128x128 tile / workgroup,
// 8 waves x (16 rows x 128 cols), K staged in LDS as bf16 (BK=32).
// =====================================================================
__launch_bounds__(256)
__global__ void qproj_ln_kernel(const float* __restrict__ x,
                                const float* __restrict__ Wq,
                                const float* __restrict__ ln_g,
                                const float* __restrict__ ln_b,
                                bf16_t* __restrict__ q_out)
{
    __shared__ __align__(16) bf16_t As[128][40];   // [m][k], 80B stride (16B aligned)
    __shared__ __align__(16) bf16_t Bs[128][40];   // [n][k]

    const int tid  = threadIdx.x;
    const int wave = tid >> 5;
    const int lane = tid & 31;
    const int l16  = lane & 15;
    const int hi   = lane >> 4;            // 0 for lanes 0-15, 1 for 16-31

    const int n0 = blockIdx.x * 128;       // output-col tile (aligned to LN group)
    const int m0 = blockIdx.y * 128;       // token tile

    v8f acc[8] = {};                       // 8 subtiles of 16x16, f32 accum

    const int ldRow = tid >> 1;            // 0..127
    const int ldK   = (tid & 1) * 16;      // 0 or 16

    for (int kk = 0; kk < DIMX; kk += 32) {
        // ---- cooperative stage: fp32 -> bf16 into LDS ----
        {
            const float4* xp = (const float4*)(x + (size_t)(m0 + ldRow) * DIMX + kk + ldK);
            float4 a0 = xp[0], a1 = xp[1], a2 = xp[2], a3 = xp[3];
            v8bf p0, p1;
            p0[0]=(bf16_t)a0.x; p0[1]=(bf16_t)a0.y; p0[2]=(bf16_t)a0.z; p0[3]=(bf16_t)a0.w;
            p0[4]=(bf16_t)a1.x; p0[5]=(bf16_t)a1.y; p0[6]=(bf16_t)a1.z; p0[7]=(bf16_t)a1.w;
            p1[0]=(bf16_t)a2.x; p1[1]=(bf16_t)a2.y; p1[2]=(bf16_t)a2.z; p1[3]=(bf16_t)a2.w;
            p1[4]=(bf16_t)a3.x; p1[5]=(bf16_t)a3.y; p1[6]=(bf16_t)a3.z; p1[7]=(bf16_t)a3.w;
            *(v8bf*)&As[ldRow][ldK]     = p0;
            *(v8bf*)&As[ldRow][ldK + 8] = p1;

            const float4* wp = (const float4*)(Wq + (size_t)(n0 + ldRow) * DIMX + kk + ldK);
            float4 b0 = wp[0], b1 = wp[1], b2 = wp[2], b3 = wp[3];
            v8bf q0, q1;
            q0[0]=(bf16_t)b0.x; q0[1]=(bf16_t)b0.y; q0[2]=(bf16_t)b0.z; q0[3]=(bf16_t)b0.w;
            q0[4]=(bf16_t)b1.x; q0[5]=(bf16_t)b1.y; q0[6]=(bf16_t)b1.z; q0[7]=(bf16_t)b1.w;
            q1[0]=(bf16_t)b2.x; q1[1]=(bf16_t)b2.y; q1[2]=(bf16_t)b2.z; q1[3]=(bf16_t)b2.w;
            q1[4]=(bf16_t)b3.x; q1[5]=(bf16_t)b3.y; q1[6]=(bf16_t)b3.z; q1[7]=(bf16_t)b3.w;
            *(v8bf*)&Bs[ldRow][ldK]     = q0;
            *(v8bf*)&Bs[ldRow][ldK + 8] = q1;
        }
        __syncthreads();

        // ---- WMMA: A frag = this wave's 16-row strip ----
        FragBF af;
        const bf16_t* arow = &As[wave * 16 + l16][0];
        af.h[0] = *(const v8bf*)(arow + hi * 8);
        af.h[1] = *(const v8bf*)(arow + 16 + hi * 8);
#pragma unroll
        for (int t = 0; t < 8; ++t) {
            FragBF bfr;
            const bf16_t* brow = &Bs[t * 16 + l16][hi * 8];
            bfr.h[0] = *(const v8bf*)(brow);
            bfr.h[1] = *(const v8bf*)(brow + 16);
            acc[t] = __builtin_amdgcn_wmma_f32_16x16x32_bf16(
                false, af.v, false, bfr.v, (short)0, acc[t], false, false);
        }
        __syncthreads();
    }

    // ---- fused LayerNorm over the 128 cols of this tile (wave-local) ----
    float gv[8], bv[8];
#pragma unroll
    for (int t = 0; t < 8; ++t) {
        int d = t * 16 + l16;              // 0..127 inside LN segment
        gv[t] = ln_g[d];
        bv[t] = ln_b[d];
    }

#pragma unroll
    for (int r = 0; r < 8; ++r) {
        // C/D layout: VGPR r holds row r (lanes 0-15) / row r+8 (lanes 16-31)
        float s = 0.f;
#pragma unroll
        for (int t = 0; t < 8; ++t) s += acc[t][r];
        s += __shfl_xor(s, 1); s += __shfl_xor(s, 2);
        s += __shfl_xor(s, 4); s += __shfl_xor(s, 8);   // stays within 16-lane half
        float mu = s * (1.0f / 128.0f);
        float q2 = 0.f;
#pragma unroll
        for (int t = 0; t < 8; ++t) { float d = acc[t][r] - mu; q2 += d * d; }
        q2 += __shfl_xor(q2, 1); q2 += __shfl_xor(q2, 2);
        q2 += __shfl_xor(q2, 4); q2 += __shfl_xor(q2, 8);
        float rs = rsqrtf(q2 * (1.0f / 128.0f) + LN_EPS);
        int row = m0 + wave * 16 + r + hi * 8;
#pragma unroll
        for (int t = 0; t < 8; ++t) {
            float val = (acc[t][r] - mu) * rs * gv[t] + bv[t];
            q_out[(size_t)row * NQ + n0 + t * 16 + l16] = (bf16_t)val;
        }
    }
}

// =====================================================================
// Kernel 2: dots[token][h*2+p][n] = q_ln[token, p,h,:] . keys[h,n,p,:]
// 16 GEMMs of (4096x128)@(128x256); grid.z = (h,p), same tiling.
// A tile is already bf16 in global -> copied with the CDNA5 async
// global->LDS engine (ASYNCcnt), no VGPR round trip.
// =====================================================================
__launch_bounds__(256)
__global__ void dots_kernel(const bf16_t* __restrict__ q,
                            const float* __restrict__ keys,
                            float* __restrict__ dots)
{
    __shared__ __align__(16) bf16_t As[128][40];
    __shared__ __align__(16) bf16_t Bs[128][40];

    const int tid  = threadIdx.x;
    const int wave = tid >> 5;
    const int lane = tid & 31;
    const int l16  = lane & 15;
    const int hi   = lane >> 4;

    const int n0 = blockIdx.x * 128;       // key tile within NK
    const int m0 = blockIdx.y * 128;       // token tile
    const int hp = blockIdx.z;             // 0..15
    const int h  = hp >> 1;
    const int p  = hp & 1;
    const int qcol0 = p * (HH * DH) + h * DH;

    v8f acc[8] = {};

    const int ldRow = tid >> 1;
    const int ldK   = (tid & 1) * 16;

    // LDS byte offset of this thread's A destination (flat->LDS: low 32 bits)
    const unsigned ldsA = (unsigned)(unsigned long long)(uintptr_t)&As[ldRow][ldK];

    for (int kk = 0; kk < DH; kk += 32) {
        // ---- A tile: async global->LDS copy (32B per thread, 2 x b128).
        // INST_OFFSET adds to both the memory and the LDS address, and the
        // two 16B chunks are contiguous in both spaces, so offset:16 works.
        {
            const bf16_t* gsrc = q + (size_t)(m0 + ldRow) * NQ + qcol0 + kk + ldK;
            asm volatile(
                "global_load_async_to_lds_b128 %0, %1, off\n\t"
                "global_load_async_to_lds_b128 %0, %1, off offset:16"
                :
                : "v"(ldsA), "v"(gsrc)
                : "memory");

            // ---- B tile: keys[h][n][p][d] fp32 -> bf16 (needs conversion)
            const float4* kp = (const float4*)(keys +
                ((size_t)(h * NKEY + n0 + ldRow) * 2 + p) * DH + kk + ldK);
            float4 b0 = kp[0], b1 = kp[1], b2 = kp[2], b3 = kp[3];
            v8bf q0, q1;
            q0[0]=(bf16_t)b0.x; q0[1]=(bf16_t)b0.y; q0[2]=(bf16_t)b0.z; q0[3]=(bf16_t)b0.w;
            q0[4]=(bf16_t)b1.x; q0[5]=(bf16_t)b1.y; q0[6]=(bf16_t)b1.z; q0[7]=(bf16_t)b1.w;
            q1[0]=(bf16_t)b2.x; q1[1]=(bf16_t)b2.y; q1[2]=(bf16_t)b2.z; q1[3]=(bf16_t)b2.w;
            q1[4]=(bf16_t)b3.x; q1[5]=(bf16_t)b3.y; q1[6]=(bf16_t)b3.z; q1[7]=(bf16_t)b3.w;
            *(v8bf*)&Bs[ldRow][ldK]     = q0;
            *(v8bf*)&Bs[ldRow][ldK + 8] = q1;
        }
        // async loads are tracked by ASYNCcnt, not DScnt: drain before barrier
        asm volatile("s_wait_asynccnt 0x0" ::: "memory");
        __syncthreads();

        FragBF af;
        const bf16_t* arow = &As[wave * 16 + l16][0];
        af.h[0] = *(const v8bf*)(arow + hi * 8);
        af.h[1] = *(const v8bf*)(arow + 16 + hi * 8);
#pragma unroll
        for (int t = 0; t < 8; ++t) {
            FragBF bfr;
            const bf16_t* brow = &Bs[t * 16 + l16][hi * 8];
            bfr.h[0] = *(const v8bf*)(brow);
            bfr.h[1] = *(const v8bf*)(brow + 16);
            acc[t] = __builtin_amdgcn_wmma_f32_16x16x32_bf16(
                false, af.v, false, bfr.v, (short)0, acc[t], false, false);
        }
        __syncthreads();
    }

    // store f32 dots: [token][hp][n]
#pragma unroll
    for (int r = 0; r < 8; ++r) {
        int row = m0 + wave * 16 + r + hi * 8;
#pragma unroll
        for (int t = 0; t < 8; ++t) {
            dots[(size_t)row * (16 * NKEY) + hp * NKEY + n0 + t * 16 + l16] = acc[t][r];
        }
    }
}

// =====================================================================
// Kernel 3: per (token, head): top-16 of each 256-dot list (p=0,1),
// 16x16 pair sums, top-16 of those, softmax. Order-invariant
// replace-min keeps exactly the top-k set (softmax/output don't need
// sorted order).
// =====================================================================
__launch_bounds__(256)
__global__ void topk_softmax_kernel(const float* __restrict__ dots,
                                    float* __restrict__ attn,
                                    int* __restrict__ vidx)
{
    const int gid   = blockIdx.x * blockDim.x + threadIdx.x;  // 0..32767
    const int token = gid >> 3;
    const int h     = gid & 7;

    const float* base0 = dots + (size_t)token * (16 * NKEY) + (h * 2 + 0) * NKEY;
    const float* base1 = base0 + NKEY;

    float sx[TOPK], sy[TOPK];
    int   ix[TOPK], iy[TOPK];
#pragma unroll
    for (int j = 0; j < TOPK; ++j) { sx[j] = -3.0e38f; sy[j] = -3.0e38f; ix[j] = 0; iy[j] = 0; }

    for (int n = 0; n < NKEY; ++n) {
        float v0 = base0[n];
        float mn = sx[0]; int ms = 0;
#pragma unroll
        for (int j = 1; j < TOPK; ++j) if (sx[j] < mn) { mn = sx[j]; ms = j; }
        if (v0 > mn) { sx[ms] = v0; ix[ms] = n; }

        float v1 = base1[n];
        float mn1 = sy[0]; int ms1 = 0;
#pragma unroll
        for (int j = 1; j < TOPK; ++j) if (sy[j] < mn1) { mn1 = sy[j]; ms1 = j; }
        if (v1 > mn1) { sy[ms1] = v1; iy[ms1] = n; }
    }

    float fv[TOPK]; int fi[TOPK];
#pragma unroll
    for (int j = 0; j < TOPK; ++j) { fv[j] = -3.0e38f; fi[j] = 0; }

    for (int i = 0; i < TOPK; ++i) {
        float sxi = sx[i];
        int   ib  = ix[i] * NKEY;
        for (int j = 0; j < TOPK; ++j) {
            float s = sxi + sy[j];
            float mn = fv[0]; int ms = 0;
#pragma unroll
            for (int u = 1; u < TOPK; ++u) if (fv[u] < mn) { mn = fv[u]; ms = u; }
            if (s > mn) { fv[ms] = s; fi[ms] = ib + iy[j]; }
        }
    }

    // softmax over the kept 16
    float mx = fv[0];
#pragma unroll
    for (int j = 1; j < TOPK; ++j) mx = fmaxf(mx, fv[j]);
    float e[TOPK], sum = 0.f;
#pragma unroll
    for (int j = 0; j < TOPK; ++j) { e[j] = __expf(fv[j] - mx); sum += e[j]; }
    float inv = 1.0f / sum;

    float* ao = attn + (size_t)token * (HH * TOPK) + h * TOPK;
    int*   io = vidx + (size_t)token * (HH * TOPK) + h * TOPK;
#pragma unroll
    for (int j = 0; j < TOPK; ++j) { ao[j] = e[j] * inv; io[j] = fi[j]; }
}

// =====================================================================
// Kernel 4: out[token,:] = sum_{h,k} attn * values[vidx,:]
// One block per token; weights/indices staged in LDS; values is
// L2-resident (128 MB < 192 MB L2).
// =====================================================================
__launch_bounds__(256)
__global__ void gather_out_kernel(const float* __restrict__ attn,
                                  const int* __restrict__ vidx,
                                  const float* __restrict__ values,
                                  float* __restrict__ out)
{
    __shared__ float w_s[HH * TOPK];
    __shared__ int   i_s[HH * TOPK];

    const int token = blockIdx.x;
    const int tid   = threadIdx.x;
    if (tid < HH * TOPK) {
        w_s[tid] = attn[(size_t)token * (HH * TOPK) + tid];
        i_s[tid] = vidx[(size_t)token * (HH * TOPK) + tid];
    }
    __syncthreads();

    const int col = tid * 2;               // 256 threads x float2 = 512 cols
    float2 acc = {0.f, 0.f};
#pragma unroll 4
    for (int e = 0; e < HH * TOPK; ++e) {
        if (e + 4 < HH * TOPK)
            __builtin_prefetch(values + (size_t)i_s[e + 4] * DIMX + col, 0, 0);
        float wt = w_s[e];
        float2 vv = *(const float2*)(values + (size_t)i_s[e] * DIMX + col);
        acc.x += wt * vv.x;
        acc.y += wt * vv.y;
    }
    *(float2*)(out + (size_t)token * DIMX + col) = acc;
}

// =====================================================================
extern "C" void kernel_launch(void* const* d_in, const int* in_sizes, int n_in,
                              void* d_out, int out_size, void* d_ws, size_t ws_size,
                              hipStream_t stream)
{
    (void)in_sizes; (void)n_in; (void)out_size; (void)ws_size;

    const float* x      = (const float*)d_in[0];
    const float* Wq     = (const float*)d_in[1];
    const float* ln_g   = (const float*)d_in[2];
    const float* ln_b   = (const float*)d_in[3];
    const float* keys   = (const float*)d_in[4];
    const float* values = (const float*)d_in[5];
    float* out = (float*)d_out;

    char* ws = (char*)d_ws;
    size_t off = 0;
    bf16_t* q_bf = (bf16_t*)(ws + off); off += (size_t)NTOK * NQ * sizeof(bf16_t);        // 16 MB
    float*  dots = (float*)(ws + off);  off += (size_t)NTOK * 16 * NKEY * sizeof(float);  // 64 MB
    float*  attn = (float*)(ws + off);  off += (size_t)NTOK * HH * TOPK * sizeof(float);  //  2 MB
    int*    vidx = (int*)(ws + off);    off += (size_t)NTOK * HH * TOPK * sizeof(int);    //  2 MB

    qproj_ln_kernel<<<dim3(NQ / 128, NTOK / 128), 256, 0, stream>>>(x, Wq, ln_g, ln_b, q_bf);
    dots_kernel<<<dim3(NKEY / 128, NTOK / 128, 16), 256, 0, stream>>>(q_bf, keys, dots);
    topk_softmax_kernel<<<dim3((NTOK * HH) / 256), 256, 0, stream>>>(dots, attn, vidx);
    gather_out_kernel<<<dim3(NTOK), 256, 0, stream>>>(attn, vidx, values, out);
}
